// Head_66614942760980
// MI455X (gfx1250) — matrix operand order repaired and checked by reading, not compiled, not run on hardware
//
#include <hip/hip_runtime.h>
#include <hip/hip_bf16.h>

// Problem constants (reference: B,T,C,H = 4,4096,512,64)
#define BDIM 4
#define TDIM 4096
#define CDIM 512
#define HDIM 64
#define QTILES (TDIM / 16)   // 256 query tiles per batch

typedef __attribute__((ext_vector_type(16))) _Float16 v16h;
typedef __attribute__((ext_vector_type(8)))  _Float16 v8h;
typedef __attribute__((ext_vector_type(8)))  float    v8f;

// ---------------------------------------------------------------------------
// A-fragment loader for V_WMMA_F32_16X16X32_F16 (16-bit A, 16x32, wave32).
// Contiguous order: lanes 0-15 hold M=0..15 with K0..15; lanes 16-31 hold the
// same rows with K16..31. Each lane reads two contiguous 8-half (16B) chunks at
// offsets k0 and k0+8 within the K window.
// ---------------------------------------------------------------------------
__device__ __forceinline__ v16h load_afrag(const _Float16* rowbase, int k0) {
  v8h lo = *(const v8h*)(rowbase + k0);        // VGPR0-3
  v8h hi = *(const v8h*)(rowbase + k0 + 8);    // VGPR4-7
  v16h r;
#pragma unroll
  for (int i = 0; i < 8; ++i) { r[i] = lo[i]; r[i + 8] = hi[i]; }
  return r;
}

// ---------------------------------------------------------------------------
// Weight transpose + f32->f16: WT[h][k] = W[k][h] for the three projections.
// 64KB each, L2-resident; makes B-fragments contiguous 32B loads.
// ---------------------------------------------------------------------------
__global__ void __launch_bounds__(256) transpose_w_kernel(
    const float* __restrict__ qW, const float* __restrict__ kW,
    const float* __restrict__ vW,
    _Float16* __restrict__ tq, _Float16* __restrict__ tk,
    _Float16* __restrict__ tv) {
  int idx = blockIdx.x * 256 + threadIdx.x;   // [0, 3*C*H)
  int which = idx >> 15;                      // / (C*H = 32768)
  int i = idx & 32767;
  int h = i >> 9;                             // / CDIM
  int k = i & (CDIM - 1);
  const float* src = (which == 0) ? qW : (which == 1) ? kW : vW;
  _Float16*    dst = (which == 0) ? tq : (which == 1) ? tk : tv;
  dst[i] = (_Float16)src[(size_t)k * HDIM + h];
}

// ---------------------------------------------------------------------------
// WMMA projection: out = (X @ W + b) * scale, f16 outputs.
// One wave per 16-row M-tile, all H=64 columns (4 accumulators), K-loop of 16
// v_wmma_f32_16x16x32_f16 x4. A: f32 global loads + cvt; B: contiguous f16
// loads from the pre-transposed WT. 8 waves / WG, 128 WGs per projection.
// ---------------------------------------------------------------------------
__global__ void __launch_bounds__(256) proj_wmma_kernel(
    const float* __restrict__ in,      // [B*T, C] f32
    const _Float16* __restrict__ WT,   // [H, C]   f16 (transposed weights)
    const float* __restrict__ bias,    // [H]      f32
    float scale,
    _Float16* __restrict__ out,        // [B*T, H] f16 or null
    _Float16* __restrict__ outT) {     // [B, H, T] f16 or null
  const int wv   = threadIdx.x >> 5;
  const int lane = threadIdx.x & 31;
  const int half = lane >> 4;
  const int col  = lane & 15;
  const int r0   = (blockIdx.x * 8 + wv) * 16;   // M-tile base row

  const float* arow = in + (size_t)(r0 + col) * CDIM;
  v8f acc[4] = {};
#pragma unroll 4
  for (int kw = 0; kw < CDIM; kw += 32) {
    // A fragment: 16 f32 -> f16 per lane (two 32B chunks at k0 and k0+8)
    const float* ap = arow + kw + half * 16;
    v16h a;
#pragma unroll
    for (int i = 0; i < 8; ++i) {
      a[i]     = (_Float16)ap[i];
      a[i + 8] = (_Float16)ap[i + 8];
    }
#pragma unroll
    for (int t = 0; t < 4; ++t) {
      // B fragment: lane holds W column h=16t+col over this K window,
      // contiguous 16 halfs (32B) at half*16 within the window.
      v16h bw = *(const v16h*)(WT + (size_t)(16 * t + col) * CDIM + kw + half * 16);
      acc[t] = __builtin_amdgcn_wmma_f32_16x16x32_f16(false, a, false, bw,
                                                      (short)0, acc[t], false, false);
    }
  }
  // bias + scale in f32, then f16 stores
#pragma unroll
  for (int t = 0; t < 4; ++t) {
    const int h = 16 * t + col;
    const float bb = bias[h];
#pragma unroll
    for (int i = 0; i < 8; ++i) {
      int r = r0 + 8 * half + i;
      _Float16 hv = (_Float16)((acc[t][i] + bb) * scale);
      if (out)  out[(size_t)r * HDIM + h] = hv;
      if (outT) outT[((size_t)(r >> 12) * HDIM + h) * TDIM + (r & (TDIM - 1))] = hv;
    }
  }
}

// ---------------------------------------------------------------------------
// Causal flash-attention forward. One wave per 16-row query tile, 4 waves/WG.
// Work-balanced tile pairing: WG x handles {2x, 2x+1, 255-2x, 254-2x} so every
// WG processes the same number of key blocks despite the causal triangle.
// ---------------------------------------------------------------------------
__global__ void __launch_bounds__(128) flash_kernel(
    const _Float16* __restrict__ qp,   // [B,T,H]  f16, pre-scaled by C^-0.5
    const _Float16* __restrict__ kp,   // [B,T,H]  f16
    const _Float16* __restrict__ vT,   // [B,H,T]  f16 (transposed V projection)
    float* __restrict__ out) {         // [B,T,H]  f32
  __shared__ _Float16 pbuf[4][16][32];  // per-wave P tile (16 q-rows x 32 keys)

  const int wv   = threadIdx.x >> 5;
  const int lane = threadIdx.x & 31;
  const int half = lane >> 4;   // 0: rows M=0..7 / K-lo, 1: rows M=8..15 / K-hi
  const int col  = lane & 15;   // N (column) in C-layout; M for frag row loads
  const int b    = blockIdx.y;
  const int qtile = (wv < 2) ? (2 * blockIdx.x + wv)
                             : ((QTILES - 1) - 2 * blockIdx.x - (wv - 2));
  const int qt16 = qtile * 16;

  // Q tile A-fragments: K window 0..31 and 32..63 of H=64
  const _Float16* qrow = qp + ((size_t)b * TDIM + qt16 + col) * HDIM;
  const v16h a0 = load_afrag(qrow,      half * 16);
  const v16h a1 = load_afrag(qrow + 32, half * 16);

  v8f acc[4] = {};       // O accumulators, 4 H-tiles of 16
  float m[8], l[8];      // per-row running max / sum (rows 8*half+i)
#pragma unroll
  for (int i = 0; i < 8; ++i) { m[i] = -1e30f; l[i] = 0.f; }

  const int kend = qt16 + 16;  // keys needed: [0, qt16+16)
  for (int kb = 0; kb < kend; kb += 32) {
    if (kb + 32 < kend)
      __builtin_prefetch(kp + ((size_t)b * TDIM + kb + 32 + col) * HDIM, 0, 0);

    // ---- S = qp_tile @ kp_tile^T : two 16-key subtiles, K=64 = 2 x 32 ----
    v8f s[2];
#pragma unroll
    for (int sub = 0; sub < 2; ++sub) {
      const _Float16* krow = kp + ((size_t)b * TDIM + kb + sub * 16 + col) * HDIM;
      v16h bk0 = *(const v16h*)(krow + half * 16);
      v16h bk1 = *(const v16h*)(krow + 32 + half * 16);
      v8f z = {};
      z = __builtin_amdgcn_wmma_f32_16x16x32_f16(false, a0, false, bk0,
                                                 (short)0, z, false, false);
      z = __builtin_amdgcn_wmma_f32_16x16x32_f16(false, a1, false, bk1,
                                                 (short)0, z, false, false);
      s[sub] = z;
    }

    // ---- causal mask + per-row block max ----
    float rmax[8];
    const int key0 = kb + col, key1 = kb + 16 + col;
#pragma unroll
    for (int i = 0; i < 8; ++i) {
      const int qi = qt16 + 8 * half + i;     // query row of element i
      float s0 = (key0 <= qi) ? s[0][i] : -1e30f;
      float s1 = (key1 <= qi) ? s[1][i] : -1e30f;
      s[0][i] = s0; s[1][i] = s1;
      rmax[i] = fmaxf(s0, s1);
    }
#pragma unroll
    for (int st = 1; st < 16; st <<= 1) {     // row lives across 16 lanes
#pragma unroll
      for (int i = 0; i < 8; ++i)
        rmax[i] = fmaxf(rmax[i], __shfl_xor(rmax[i], st, 32));
    }

    // ---- online softmax update, stage P (f16) into LDS in C-layout ----
#pragma unroll
    for (int i = 0; i < 8; ++i) {
      float nm   = fmaxf(m[i], rmax[i]);
      float corr = __expf(m[i] - nm);
      m[i] = nm;
      float p0 = __expf(s[0][i] - nm);
      float p1 = __expf(s[1][i] - nm);
      l[i] = l[i] * corr + p0 + p1;           // per-lane partial row sum
#pragma unroll
      for (int t = 0; t < 4; ++t) acc[t][i] *= corr;
      pbuf[wv][8 * half + i][col]      = (_Float16)p0;
      pbuf[wv][8 * half + i][16 + col] = (_Float16)p1;
    }

    // ---- O += P (16x32) @ V (32x64): reload P as A-fragment from LDS ----
    // (same-wave DS ops are in-order, no barrier needed)
    v16h ap = load_afrag(&pbuf[wv][col][0], half * 16);
#pragma unroll
    for (int t = 0; t < 4; ++t) {
      const _Float16* vrow = vT + ((size_t)b * HDIM + 16 * t + col) * TDIM + kb;
      v16h bv = *(const v16h*)(vrow + half * 16);
      acc[t] = __builtin_amdgcn_wmma_f32_16x16x32_f16(false, ap, false, bv,
                                                      (short)0, acc[t], false, false);
    }
  }

  // ---- finalize: reduce row sums across the 16 lanes, normalize, store ----
#pragma unroll
  for (int st = 1; st < 16; st <<= 1) {
#pragma unroll
    for (int i = 0; i < 8; ++i)
      l[i] += __shfl_xor(l[i], st, 32);
  }
#pragma unroll
  for (int t = 0; t < 4; ++t) {
#pragma unroll
    for (int i = 0; i < 8; ++i)
      out[((size_t)b * TDIM + qt16 + 8 * half + i) * HDIM + 16 * t + col] =
          acc[t][i] / l[i];
  }
}

// ---------------------------------------------------------------------------
// Inputs (setup_inputs order): k, v, q, mask, kW, kb, vW, vb, qW, qb
// Workspace: qp | kp | vT (2MB f16 each) + 3 transposed weight mats (64KB each)
// ---------------------------------------------------------------------------
extern "C" void kernel_launch(void* const* d_in, const int* in_sizes, int n_in,
                              void* d_out, int out_size, void* d_ws, size_t ws_size,
                              hipStream_t stream) {
  const float* k  = (const float*)d_in[0];
  const float* v  = (const float*)d_in[1];
  const float* q  = (const float*)d_in[2];
  // d_in[3] = causal mask, implicit in the kernel
  const float* kW = (const float*)d_in[4];
  const float* kb = (const float*)d_in[5];
  const float* vW = (const float*)d_in[6];
  const float* vb = (const float*)d_in[7];
  const float* qW = (const float*)d_in[8];
  const float* qb = (const float*)d_in[9];
  float* out = (float*)d_out;

  const size_t NP = (size_t)BDIM * TDIM * HDIM;  // 1,048,576 elems per proj
  const size_t NW = (size_t)CDIM * HDIM;         // 32,768 elems per weight
  _Float16* qp  = (_Float16*)d_ws;
  _Float16* kp  = qp + NP;
  _Float16* vT  = kp + NP;
  _Float16* wTq = vT + NP;
  _Float16* wTk = wTq + NW;
  _Float16* wTv = wTk + NW;

  const float scale = 0.04419417382415922f;  // C^-0.5 = 512^-0.5, folded into qp

  transpose_w_kernel<<<dim3((unsigned)(3 * NW / 256)), dim3(256), 0, stream>>>(
      qW, kW, vW, wTq, wTk, wTv);

  dim3 pgrid((unsigned)(BDIM * TDIM / 16 / 8)), pblk(256);  // 128 WGs, 8 waves
  proj_wmma_kernel<<<pgrid, pblk, 0, stream>>>(q, wTq, qb, scale, qp, nullptr);
  proj_wmma_kernel<<<pgrid, pblk, 0, stream>>>(k, wTk, kb, 1.0f,  kp, nullptr);
  proj_wmma_kernel<<<pgrid, pblk, 0, stream>>>(v, wTv, vb, 1.0f,  nullptr, vT);

  dim3 fgrid(QTILES / 4, BDIM), fblk(128);   // 4 waves/WG, balanced tile pairs
  flash_kernel<<<fgrid, fblk, 0, stream>>>(qp, kp, vT, out);
}
